// Moirai2Module_83099027243787
// MI455X (gfx1250) — compile-verified, matrix operand order checked by reading, not run on hardware
//
#include <hip/hip_runtime.h>
#include <hip/hip_bf16.h>
#include <math.h>

// ---------------------------------------------------------------------------
// Types for CDNA5 WMMA (wave32): v_wmma_f32_16x16x32_bf16
// ---------------------------------------------------------------------------
typedef __bf16 bf16;
typedef __attribute__((ext_vector_type(8)))  bf16  bf16x8;
typedef __attribute__((ext_vector_type(16))) bf16  v16bf;
typedef __attribute__((ext_vector_type(8)))  float v8f;

union V16 { v16bf v; bf16x8 h[2]; };

static constexpr int   B_    = 2;
static constexpr int   S_    = 2048;
static constexpr int   M_    = B_ * S_;      // 4096 tokens
static constexpr int   DM    = 384;
static constexpr int   DFF   = 1024;
static constexpr int   H_    = 6;
static constexpr int   HD    = 64;
static constexpr int   PATCH = 16;
static constexpr int   ODIM  = 144;          // 9*PATCH
static constexpr float EPS_  = 1e-5f;
static constexpr float MIN_SCALE = 1e-5f;

// ---------------------------------------------------------------------------
// fp32 -> bf16 conversion
// ---------------------------------------------------------------------------
__global__ void cvt_f32_bf16(const float* __restrict__ in, bf16* __restrict__ out, int n) {
  int i = blockIdx.x * blockDim.x + threadIdx.x;
  if (i < n) out[i] = (bf16)in[i];
}

// ---------------------------------------------------------------------------
// Std scaler kernels
// ---------------------------------------------------------------------------
__global__ void scaler_token_sums(const float* __restrict__ target,
                                  const unsigned char* __restrict__ omask,
                                  const unsigned char* __restrict__ pmask,
                                  float* __restrict__ tobs, float* __restrict__ tsum) {
  int t = blockIdx.x * blockDim.x + threadIdx.x;
  if (t >= M_) return;
  float o = 0.f, s = 0.f;
  const bool pm = pmask[t] != 0;
  for (int p = 0; p < PATCH; ++p) {
    float ob = (omask[t * PATCH + p] && !pm) ? 1.f : 0.f;
    o += ob;
    s += target[t * PATCH + p] * ob;
  }
  tobs[t] = o; tsum[t] = s;
}

__global__ void scaler_loc(const int* __restrict__ sid, const int* __restrict__ vid,
                           const float* __restrict__ tobs, const float* __restrict__ tsum,
                           float* __restrict__ locv, float* __restrict__ tobsg) {
  __shared__ float s1[256], s2[256];
  const int t = blockIdx.x;
  const int b = t / S_;
  const int si = sid[t], vi = vid[t];
  float a = 0.f, c = 0.f;
  for (int j = threadIdx.x; j < S_; j += 256) {
    int g = b * S_ + j;
    if (sid[g] == si && vid[g] == vi) { a += tobs[g]; c += tsum[g]; }
  }
  s1[threadIdx.x] = a; s2[threadIdx.x] = c;
  __syncthreads();
  for (int st = 128; st; st >>= 1) {
    if (threadIdx.x < st) { s1[threadIdx.x] += s1[threadIdx.x + st]; s2[threadIdx.x] += s2[threadIdx.x + st]; }
    __syncthreads();
  }
  if (threadIdx.x == 0) {
    float tg = s1[0];
    locv[t]  = s2[0] / fmaxf(tg, 1.f);
    tobsg[t] = tg;
  }
}

__global__ void scaler_vsum(const float* __restrict__ target,
                            const unsigned char* __restrict__ omask,
                            const unsigned char* __restrict__ pmask,
                            const float* __restrict__ locv, float* __restrict__ vsum) {
  int t = blockIdx.x * blockDim.x + threadIdx.x;
  if (t >= M_) return;
  float l = locv[t], vs = 0.f;
  const bool pm = pmask[t] != 0;
  for (int p = 0; p < PATCH; ++p) {
    float ob = (omask[t * PATCH + p] && !pm) ? 1.f : 0.f;
    float d = target[t * PATCH + p] - l;
    vs += d * d * ob;
  }
  vsum[t] = vs;
}

__global__ void scaler_finish(const float* __restrict__ target,
                              const unsigned char* __restrict__ omask,
                              const int* __restrict__ sid, const int* __restrict__ vid,
                              const float* __restrict__ vsum, const float* __restrict__ locv,
                              const float* __restrict__ tobsg,
                              float* __restrict__ scaled_out, bf16* __restrict__ tokens) {
  __shared__ float s1[256];
  const int t = blockIdx.x;
  const int b = t / S_;
  const int si = sid[t], vi = vid[t];
  float a = 0.f;
  for (int j = threadIdx.x; j < S_; j += 256) {
    int g = b * S_ + j;
    if (sid[g] == si && vid[g] == vi) a += vsum[g];
  }
  s1[threadIdx.x] = a;
  __syncthreads();
  for (int st = 128; st; st >>= 1) {
    if (threadIdx.x < st) s1[threadIdx.x] += s1[threadIdx.x + st];
    __syncthreads();
  }
  if (threadIdx.x < PATCH) {
    float var   = s1[0] / fmaxf(tobsg[t] - 1.f, 1.f);
    float scale = sqrtf(var + MIN_SCALE);
    float loc   = locv[t];
    if (si == 0) { loc = 0.f; scale = 1.f; }   // pad handling
    int p = threadIdx.x;
    float ob = omask[t * PATCH + p] ? 1.f : 0.f;
    float sc = (ob != 0.f) ? (target[t * PATCH + p] - loc) / scale : 0.f;
    scaled_out[t * PATCH + p]   = sc;
    tokens[t * 32 + p]          = (bf16)sc;
    tokens[t * 32 + PATCH + p]  = (bf16)ob;
  }
}

// ---------------------------------------------------------------------------
// WMMA GEMM, 32(M) x 64(N) register-blocked tile per wave.
// C = A_bf16(MxK) @ W_bf16(NxK)^T (+bias) (+= opt).  K%32==0, M%32==0, N%64==0
// 8 WMMAs per K-step on 12 b128 loads (A fragments reused across 4 N-tiles).
// __launch_bounds__(32,1): single wave per group, allow full VGPR budget
// (round-2 build spilled 24B of address regs in the K-loop without this).
// ---------------------------------------------------------------------------
__global__ __launch_bounds__(32, 1)
void gemm_wmma_32x64(const bf16* __restrict__ A, const bf16* __restrict__ W,
                     const float* __restrict__ bias, float* __restrict__ C,
                     int M, int N, int K, int accumulate) {
  const int lane = threadIdx.x;
  const int m0 = blockIdx.y * 32;
  const int n0 = blockIdx.x * 64;
  const int r16  = lane & 15;
  const int base = (lane < 16) ? 0 : 8;
  const bf16* arow0 = A + (size_t)(m0 + r16) * K;
  const bf16* arow1 = A + (size_t)(m0 + 16 + r16) * K;
  const bf16* wrow[4];
#pragma unroll
  for (int j = 0; j < 4; ++j) wrow[j] = W + (size_t)(n0 + j * 16 + r16) * K;
  v8f acc[2][4] = {{{}, {}, {}, {}}, {{}, {}, {}, {}}};
  for (int k0 = 0; k0 < K; k0 += 32) {
    __builtin_prefetch(arow0 + k0 + 64, 0, 1);
    __builtin_prefetch(arow1 + k0 + 64, 0, 1);
    V16 a0, a1;
    a0.h[0] = *(const bf16x8*)(arow0 + k0 + base);
    a0.h[1] = *(const bf16x8*)(arow0 + k0 + 16 + base);
    a1.h[0] = *(const bf16x8*)(arow1 + k0 + base);
    a1.h[1] = *(const bf16x8*)(arow1 + k0 + 16 + base);
#pragma unroll
    for (int j = 0; j < 4; ++j) {
      V16 w;
      w.h[0] = *(const bf16x8*)(wrow[j] + k0 + base);
      w.h[1] = *(const bf16x8*)(wrow[j] + k0 + 16 + base);
      acc[0][j] = __builtin_amdgcn_wmma_f32_16x16x32_bf16(false, a0.v, false, w.v,
                                                          (short)0, acc[0][j], false, false);
      acc[1][j] = __builtin_amdgcn_wmma_f32_16x16x32_bf16(false, a1.v, false, w.v,
                                                          (short)0, acc[1][j], false, false);
    }
  }
  const int hi8 = (lane >> 4) << 3;
#pragma unroll
  for (int j = 0; j < 4; ++j) {
    const int n = n0 + j * 16 + r16;
    const float bb = bias ? bias[n] : 0.0f;
#pragma unroll
    for (int i = 0; i < 2; ++i)
#pragma unroll
      for (int r = 0; r < 8; ++r) {
        const int m = m0 + i * 16 + r + hi8;
        float v = acc[i][j][r] + bb;
        if (accumulate) C[(size_t)m * N + n] += v;
        else            C[(size_t)m * N + n]  = v;
      }
  }
}

// ---------------------------------------------------------------------------
// WMMA GEMM, 16x16 tile per wave (for N=144 output GEMMs)
// ---------------------------------------------------------------------------
__global__ __launch_bounds__(32, 1)
void gemm_wmma_16(const bf16* __restrict__ A, const bf16* __restrict__ W,
                  const float* __restrict__ bias, float* __restrict__ C,
                  int M, int N, int K, int accumulate) {
  const int lane = threadIdx.x;
  const int m0 = blockIdx.y * 16;
  const int n0 = blockIdx.x * 16;
  const int r16  = lane & 15;
  const int base = (lane < 16) ? 0 : 8;
  const bf16* arow = A + (size_t)(m0 + r16) * K;
  const bf16* wrow = W + (size_t)(n0 + r16) * K;
  v8f acc = {};
  for (int k0 = 0; k0 < K; k0 += 32) {
    __builtin_prefetch(arow + k0 + 64, 0, 1);
    __builtin_prefetch(wrow + k0 + 64, 0, 1);
    V16 a, b;
    a.h[0] = *(const bf16x8*)(arow + k0 + base);
    a.h[1] = *(const bf16x8*)(arow + k0 + 16 + base);
    b.h[0] = *(const bf16x8*)(wrow + k0 + base);
    b.h[1] = *(const bf16x8*)(wrow + k0 + 16 + base);
    acc = __builtin_amdgcn_wmma_f32_16x16x32_bf16(false, a.v, false, b.v,
                                                  (short)0, acc, false, false);
  }
  const int hi8 = (lane >> 4) << 3;
  const int n = n0 + r16;
  const float bb = bias ? bias[n] : 0.0f;
#pragma unroll
  for (int r = 0; r < 8; ++r) {
    const int m = m0 + r + hi8;
    float v = acc[r] + bb;
    if (accumulate) C[(size_t)m * N + n] += v;
    else            C[(size_t)m * N + n]  = v;
  }
}

// ---------------------------------------------------------------------------
// RMSNorm over D_MODEL, fp32 in -> bf16 out; one wave per token
// ---------------------------------------------------------------------------
__global__ void rmsnorm_bf16(const float* __restrict__ X, const float* __restrict__ w,
                             bf16* __restrict__ Hout) {
  const int t = blockIdx.x;
  const int lane = threadIdx.x;
  const float* x = X + (size_t)t * DM;
  float ss = 0.f;
  for (int i = lane; i < DM; i += 32) { float v = x[i]; ss += v * v; }
  for (int msk = 16; msk; msk >>= 1) ss += __shfl_xor(ss, msk, 32);
  const float rs = rsqrtf(ss / DM + EPS_);
  bf16* o = Hout + (size_t)t * DM;
  for (int i = lane; i < DM; i += 32) o[i] = (bf16)(x[i] * rs * w[i]);
}

// ---------------------------------------------------------------------------
// Per-head RMSNorm (64) + RoPE(32) for q/k; write (b,h,s,64) bf16 layout
// grid: (H, M), block 32; lane owns dims {lane, lane+32}
// ---------------------------------------------------------------------------
__global__ void head_norm_rope(const float* __restrict__ QK, const float* __restrict__ w,
                               const int* __restrict__ time_id, bf16* __restrict__ Out) {
  const int h = blockIdx.x;
  const int t = blockIdx.y;
  const int b = t / S_, s = t % S_;
  const int lane = threadIdx.x;
  const float* in = QK + (size_t)t * DM + h * HD;
  float x0 = in[lane], x1 = in[lane + 32];
  float ss = x0 * x0 + x1 * x1;
  for (int msk = 16; msk; msk >>= 1) ss += __shfl_xor(ss, msk, 32);
  const float rs = rsqrtf(ss / HD + EPS_);
  float n0 = x0 * rs * w[lane];
  float n1 = x1 * rs * w[lane + 32];
  // RoPE on dims [0,32): even d: x*c - x_{d+1}*s ; odd d: x*c + x_{d-1}*s
  const float tt = (float)time_id[t];
  const int   i  = lane >> 1;
  const float theta = __powf(10000.0f, -((float)(2 * i)) / 32.0f);
  const float ang = tt * theta;
  const float c = __cosf(ang), sn = __sinf(ang);
  const float partner = __shfl_xor(n0, 1, 32);
  const float r0 = (lane & 1) ? (n0 * c + partner * sn) : (n0 * c - partner * sn);
  bf16* o = Out + ((size_t)(b * H_ + h) * S_ + s) * HD;
  o[lane]      = (bf16)r0;
  o[lane + 32] = (bf16)n1;
}

// pack V fp32 (B,S,384) -> bf16 (b,h,s,64)
__global__ void v_pack(const float* __restrict__ V, bf16* __restrict__ Out, int n) {
  int i = blockIdx.x * blockDim.x + threadIdx.x;
  if (i >= n) return;
  int t = i / DM, c = i % DM;
  int b = t / S_, s = t % S_;
  int h = c / HD, d = c % HD;
  Out[((size_t)(b * H_ + h) * S_ + s) * HD + d] = (bf16)V[i];
}

// ---------------------------------------------------------------------------
// Flash attention, one wave per (b, h, 16-query tile). Key tiles of 32.
// V tile staged memory->LDS with CDNA5 async loads (ASYNCcnt-tracked).
// ---------------------------------------------------------------------------
__global__ __launch_bounds__(32, 1)
void attention_wmma(const bf16* __restrict__ Q, const bf16* __restrict__ K,
                    const bf16* __restrict__ Vt,
                    const int* __restrict__ sample_id,
                    const int* __restrict__ time_id,
                    const int* __restrict__ variate_id,
                    const float* __restrict__ bias2h, // [2][H]
                    bf16* __restrict__ Obf) {
  __shared__ __align__(16) bf16 pbuf[16 * 32];
  __shared__ __align__(16) bf16 vbuf[32 * HD];
  const int lane = threadIdx.x;
  const int qt = blockIdx.x, h = blockIdx.y, b = blockIdx.z;
  const int r16  = lane & 15;
  const int base = (lane < 16) ? 0 : 8;
  const int hi8  = (lane >> 4) << 3;
  const size_t bh = (size_t)(b * H_ + h) * S_;
  const int s0 = qt * 16;

  // Q fragments: two K-halves of HEAD_DIM (k0 = 0 and 32)
  const bf16* qrow = Q + (bh + s0 + r16) * HD;
  V16 aq0, aq1;
  aq0.h[0] = *(const bf16x8*)(qrow + base);
  aq0.h[1] = *(const bf16x8*)(qrow + 16 + base);
  aq1.h[0] = *(const bf16x8*)(qrow + 32 + base);
  aq1.h[1] = *(const bf16x8*)(qrow + 48 + base);

  int qs[8], qtm[8], qv[8];
#pragma unroll
  for (int r = 0; r < 8; ++r) {
    int row = b * S_ + s0 + r + hi8;
    qs[r] = sample_id[row]; qtm[r] = time_id[row]; qv[r] = variate_id[row];
  }
  const float b0 = bias2h[h], b1 = bias2h[H_ + h];
  const float inv = 0.125f;  // 1/sqrt(64)

  v8f oacc[4] = {{}, {}, {}, {}};
  float rm[8], rl[8];
#pragma unroll
  for (int r = 0; r < 8; ++r) { rm[r] = -1e30f; rl[r] = 0.f; }

  const unsigned vbuf_lds = (unsigned)(uintptr_t)(vbuf + lane * HD);

  for (int kt = 0; kt < S_ / 32; ++kt) {
    const int t0 = kt * 32;

    // kick off async V-tile staging (32 keys x 64) memory -> LDS
    {
      const bf16* vrow = Vt + (bh + t0 + lane) * HD;
      unsigned long long ga = (unsigned long long)(uintptr_t)vrow;
#pragma unroll
      for (int e = 0; e < HD; e += 8) {
        unsigned l = vbuf_lds + (unsigned)(e * 2);
        unsigned long long g = ga + (unsigned long long)(e * 2);
        asm volatile("global_load_async_to_lds_b128 %0, %1, off"
                     :: "v"(l), "v"(g) : "memory");
      }
    }

    // scores for two key-halves
    v8f sf[2];
#pragma unroll
    for (int j = 0; j < 2; ++j) {
      const bf16* krow = K + (bh + t0 + j * 16 + r16) * HD;
      V16 kb0, kb1;
      kb0.h[0] = *(const bf16x8*)(krow + base);
      kb0.h[1] = *(const bf16x8*)(krow + 16 + base);
      kb1.h[0] = *(const bf16x8*)(krow + 32 + base);
      kb1.h[1] = *(const bf16x8*)(krow + 48 + base);
      v8f s = {};
      s = __builtin_amdgcn_wmma_f32_16x16x32_bf16(false, aq0.v, false, kb0.v, (short)0, s, false, false);
      s = __builtin_amdgcn_wmma_f32_16x16x32_bf16(false, aq1.v, false, kb1.v, (short)0, s, false, false);
      sf[j] = s;
    }
    int ks[2], ktm[2], kv[2];
#pragma unroll
    for (int j = 0; j < 2; ++j) {
      int col = b * S_ + t0 + j * 16 + r16;
      ks[j] = sample_id[col]; ktm[j] = time_id[col]; kv[j] = variate_id[col];
    }
    float val[2][8];
#pragma unroll
    for (int j = 0; j < 2; ++j)
#pragma unroll
      for (int r = 0; r < 8; ++r) {
        float v = sf[j][r] * inv + ((kv[j] == qv[r]) ? b1 : b0);
        bool ok = (ks[j] == qs[r]) && (ktm[j] <= qtm[r]);
        val[j][r] = ok ? v : -1e30f;
      }
    // online softmax per row; row stats replicated across half-wave lanes
#pragma unroll
    for (int r = 0; r < 8; ++r) {
      float lm = fmaxf(val[0][r], val[1][r]);
      for (int msk = 1; msk < 16; msk <<= 1) lm = fmaxf(lm, __shfl_xor(lm, msk, 32));
      float nm = fmaxf(rm[r], lm);
      float alpha = __expf(rm[r] - nm);
      float p0 = (val[0][r] > -1e29f) ? __expf(val[0][r] - nm) : 0.f;
      float p1 = (val[1][r] > -1e29f) ? __expf(val[1][r] - nm) : 0.f;
      float ps = p0 + p1;
      for (int msk = 1; msk < 16; msk <<= 1) ps += __shfl_xor(ps, msk, 32);
      rl[r] = rl[r] * alpha + ps;
      rm[r] = nm;
#pragma unroll
      for (int c = 0; c < 4; ++c) oacc[c][r] *= alpha;
      pbuf[(r + hi8) * 32 + r16]      = (bf16)p0;
      pbuf[(r + hi8) * 32 + 16 + r16] = (bf16)p1;
    }
    // wait for async V staging, then synchronize LDS visibility
    asm volatile("s_wait_asynccnt 0x0" ::: "memory");
    __syncthreads();
    // P fragment (A-layout) from LDS
    V16 pa;
    pa.h[0] = *(const bf16x8*)(pbuf + r16 * 32 + base);
    pa.h[1] = *(const bf16x8*)(pbuf + r16 * 32 + 16 + base);
    // O += P @ V   (four 16-wide d-chunks)
#pragma unroll
    for (int c = 0; c < 4; ++c) {
      __attribute__((aligned(32))) bf16 tmp[16];
      const int col = c * 16 + r16;
#pragma unroll
      for (int e = 0; e < 8; ++e) tmp[e]     = vbuf[(base + e) * HD + col];
#pragma unroll
      for (int e = 0; e < 8; ++e) tmp[8 + e] = vbuf[(16 + base + e) * HD + col];
      V16 vb; vb.v = *(const v16bf*)tmp;
      oacc[c] = __builtin_amdgcn_wmma_f32_16x16x32_bf16(false, pa.v, false, vb.v,
                                                        (short)0, oacc[c], false, false);
    }
    __syncthreads();
  }
  // normalize and write (B,S,384) bf16
#pragma unroll
  for (int c = 0; c < 4; ++c)
#pragma unroll
    for (int r = 0; r < 8; ++r) {
      const int row = s0 + r + hi8;
      const int d   = c * 16 + r16;
      float v = (rl[r] > 0.f) ? oacc[c][r] / rl[r] : 0.f;
      Obf[((size_t)(b * S_ + row)) * DM + h * HD + d] = (bf16)v;
    }
}

// ---------------------------------------------------------------------------
// elementwise: relu fp32 -> bf16 ; silu(a)*g -> bf16
// ---------------------------------------------------------------------------
__global__ void relu_to_bf16(const float* __restrict__ in, bf16* __restrict__ out, int n) {
  int i = blockIdx.x * blockDim.x + threadIdx.x;
  if (i < n) out[i] = (bf16)fmaxf(in[i], 0.f);
}
__global__ void silu_gate_to_bf16(const float* __restrict__ a, const float* __restrict__ g,
                                  bf16* __restrict__ out, int n) {
  int i = blockIdx.x * blockDim.x + threadIdx.x;
  if (i < n) {
    float x = a[i];
    float s = x / (1.f + __expf(-x));
    out[i] = (bf16)(s * g[i]);
  }
}

// ---------------------------------------------------------------------------
// Host orchestration
// ---------------------------------------------------------------------------
extern "C" void kernel_launch(void* const* d_in, const int* in_sizes, int n_in,
                              void* d_out, int out_size, void* d_ws, size_t ws_size,
                              hipStream_t stream) {
  (void)in_sizes; (void)n_in; (void)out_size; (void)ws_size;
  const float*         target = (const float*)d_in[0];
  const unsigned char* omask  = (const unsigned char*)d_in[1];
  const int*           sid    = (const int*)d_in[2];
  const int*           tidp   = (const int*)d_in[3];
  const int*           vidp   = (const int*)d_in[4];
  const unsigned char* pmask  = (const unsigned char*)d_in[5];
  const float* in_w1 = (const float*)d_in[6];
  const float* in_b1 = (const float*)d_in[7];
  const float* in_w2 = (const float*)d_in[8];
  const float* in_b2 = (const float*)d_in[9];
  const float* in_wr = (const float*)d_in[10];
  const float* in_br = (const float*)d_in[11];
  const float* norm1_w = (const float*)d_in[12];
  const float* Wq = (const float*)d_in[13];
  const float* Wk = (const float*)d_in[14];
  const float* Wv = (const float*)d_in[15];
  const float* qn_w = (const float*)d_in[16];
  const float* kn_w = (const float*)d_in[17];
  const float* bias_emb = (const float*)d_in[18];
  const float* Wo = (const float*)d_in[19];
  const float* norm2_w = (const float*)d_in[20];
  const float* Wff1 = (const float*)d_in[21];
  const float* Wgate = (const float*)d_in[22];
  const float* Wff2 = (const float*)d_in[23];
  const float* normf_w = (const float*)d_in[24];
  const float* out_w1 = (const float*)d_in[25];
  const float* out_b1 = (const float*)d_in[26];
  const float* out_w2 = (const float*)d_in[27];
  const float* out_b2 = (const float*)d_in[28];
  const float* out_wr = (const float*)d_in[29];
  const float* out_br = (const float*)d_in[30];

  float* out = (float*)d_out;
  float* preds_out  = out;                       // (B,S,144)
  float* scaled_out = out + (size_t)M_ * ODIM;   // (B,S,16)

  // workspace allocator
  char* ws = (char*)d_ws;
  size_t off = 0;
  auto alloc = [&](size_t bytes) -> void* {
    off = (off + 255) & ~(size_t)255;
    void* p = ws + off;
    off += bytes;
    return p;
  };

  // bf16 weight buffers
  bf16* wb_in_w1 = (bf16*)alloc((size_t)DM * 32 * 2);
  bf16* wb_in_w2 = (bf16*)alloc((size_t)DM * DM * 2);
  bf16* wb_in_wr = (bf16*)alloc((size_t)DM * 32 * 2);
  bf16* wb_Wq   = (bf16*)alloc((size_t)6 * DM * DM * 2);
  bf16* wb_Wk   = (bf16*)alloc((size_t)6 * DM * DM * 2);
  bf16* wb_Wv   = (bf16*)alloc((size_t)6 * DM * DM * 2);
  bf16* wb_Wo   = (bf16*)alloc((size_t)6 * DM * DM * 2);
  bf16* wb_Wff1 = (bf16*)alloc((size_t)6 * DFF * DM * 2);
  bf16* wb_Wg   = (bf16*)alloc((size_t)6 * DFF * DM * 2);
  bf16* wb_Wff2 = (bf16*)alloc((size_t)6 * DM * DFF * 2);
  bf16* wb_ow1  = (bf16*)alloc((size_t)DM * DM * 2);
  bf16* wb_ow2  = (bf16*)alloc((size_t)ODIM * DM * 2);
  bf16* wb_owr  = (bf16*)alloc((size_t)ODIM * DM * 2);

  // activations / scratch
  float* tobs  = (float*)alloc((size_t)M_ * 4);
  float* tsum  = (float*)alloc((size_t)M_ * 4);
  float* locv  = (float*)alloc((size_t)M_ * 4);
  float* tobsg = (float*)alloc((size_t)M_ * 4);
  float* vsum  = (float*)alloc((size_t)M_ * 4);
  bf16*  tokens = (bf16*)alloc((size_t)M_ * 32 * 2);
  float* x    = (float*)alloc((size_t)M_ * DM * 4);
  bf16*  hbf  = (bf16*)alloc((size_t)M_ * DM * 2);
  float* qf   = (float*)alloc((size_t)M_ * DM * 4);
  float* kf   = (float*)alloc((size_t)M_ * DM * 4);
  float* vf   = (float*)alloc((size_t)M_ * DM * 4);
  bf16*  qa   = (bf16*)alloc((size_t)M_ * DM * 2);
  bf16*  ka   = (bf16*)alloc((size_t)M_ * DM * 2);
  bf16*  va   = (bf16*)alloc((size_t)M_ * DM * 2);
  bf16*  obf  = (bf16*)alloc((size_t)M_ * DM * 2);
  float* ff1  = (float*)alloc((size_t)M_ * DFF * 4);
  float* gate = (float*)alloc((size_t)M_ * DFF * 4);
  bf16*  actb = (bf16*)alloc((size_t)M_ * DFF * 2);
  float* tmpf = (float*)alloc((size_t)M_ * DM * 4);
  bf16*  tbf  = (bf16*)alloc((size_t)M_ * DM * 2);

  auto cvt = [&](const float* src, bf16* dst, int n) {
    cvt_f32_bf16<<<(n + 255) / 256, 256, 0, stream>>>(src, dst, n);
  };
  auto gemm64 = [&](const bf16* A, const bf16* W, const float* bias, float* C,
                    int M, int N, int K, int acc) {
    gemm_wmma_32x64<<<dim3(N / 64, M / 32), 32, 0, stream>>>(A, W, bias, C, M, N, K, acc);
  };
  auto gemm16 = [&](const bf16* A, const bf16* W, const float* bias, float* C,
                    int M, int N, int K, int acc) {
    gemm_wmma_16<<<dim3(N / 16, M / 16), 32, 0, stream>>>(A, W, bias, C, M, N, K, acc);
  };

  // weight conversion
  cvt(in_w1, wb_in_w1, DM * 32);
  cvt(in_w2, wb_in_w2, DM * DM);
  cvt(in_wr, wb_in_wr, DM * 32);
  cvt(Wq,   wb_Wq,   6 * DM * DM);
  cvt(Wk,   wb_Wk,   6 * DM * DM);
  cvt(Wv,   wb_Wv,   6 * DM * DM);
  cvt(Wo,   wb_Wo,   6 * DM * DM);
  cvt(Wff1, wb_Wff1, 6 * DFF * DM);
  cvt(Wgate, wb_Wg,  6 * DFF * DM);
  cvt(Wff2, wb_Wff2, 6 * DM * DFF);
  cvt(out_w1, wb_ow1, DM * DM);
  cvt(out_w2, wb_ow2, ODIM * DM);
  cvt(out_wr, wb_owr, ODIM * DM);

  // std scaler + tokens + scaled output
  scaler_token_sums<<<(M_ + 255) / 256, 256, 0, stream>>>(target, omask, pmask, tobs, tsum);
  scaler_loc<<<M_, 256, 0, stream>>>(sid, vidp, tobs, tsum, locv, tobsg);
  scaler_vsum<<<(M_ + 255) / 256, 256, 0, stream>>>(target, omask, pmask, locv, vsum);
  scaler_finish<<<M_, 256, 0, stream>>>(target, omask, sid, vidp, vsum, locv, tobsg,
                                        scaled_out, tokens);

  // input residual block: x = relu(tok@w1^T+b1)@w2^T+b2 + tok@wr^T+br
  gemm64(tokens, wb_in_w1, in_b1, tmpf, M_, DM, 32, 0);
  relu_to_bf16<<<(M_ * DM + 255) / 256, 256, 0, stream>>>(tmpf, tbf, M_ * DM);
  gemm64(tbf, wb_in_w2, in_b2, x, M_, DM, DM, 0);
  gemm64(tokens, wb_in_wr, in_br, x, M_, DM, 32, 1);

  // transformer layers
  for (int l = 0; l < 6; ++l) {
    rmsnorm_bf16<<<M_, 32, 0, stream>>>(x, norm1_w + (size_t)l * DM, hbf);
    gemm64(hbf, wb_Wq + (size_t)l * DM * DM, nullptr, qf, M_, DM, DM, 0);
    gemm64(hbf, wb_Wk + (size_t)l * DM * DM, nullptr, kf, M_, DM, DM, 0);
    gemm64(hbf, wb_Wv + (size_t)l * DM * DM, nullptr, vf, M_, DM, DM, 0);
    head_norm_rope<<<dim3(H_, M_), 32, 0, stream>>>(qf, qn_w + (size_t)l * HD, tidp, qa);
    head_norm_rope<<<dim3(H_, M_), 32, 0, stream>>>(kf, kn_w + (size_t)l * HD, tidp, ka);
    v_pack<<<(M_ * DM + 255) / 256, 256, 0, stream>>>(vf, va, M_ * DM);
    attention_wmma<<<dim3(S_ / 16, H_, B_), 32, 0, stream>>>(
        qa, ka, va, sid, tidp, vidp, bias_emb + (size_t)l * 2 * H_, obf);
    gemm64(obf, wb_Wo + (size_t)l * DM * DM, nullptr, x, M_, DM, DM, 1);
    rmsnorm_bf16<<<M_, 32, 0, stream>>>(x, norm2_w + (size_t)l * DM, hbf);
    gemm64(hbf, wb_Wff1 + (size_t)l * DFF * DM, nullptr, ff1, M_, DFF, DM, 0);
    gemm64(hbf, wb_Wg  + (size_t)l * DFF * DM, nullptr, gate, M_, DFF, DM, 0);
    silu_gate_to_bf16<<<(M_ * DFF + 255) / 256, 256, 0, stream>>>(ff1, gate, actb, M_ * DFF);
    gemm64(actb, wb_Wff2 + (size_t)l * DM * DFF, nullptr, x, M_, DM, DFF, 1);
  }

  // final norm + output residual block
  rmsnorm_bf16<<<M_, 32, 0, stream>>>(x, normf_w, hbf);
  gemm64(hbf, wb_ow1, out_b1, tmpf, M_, DM, DM, 0);
  relu_to_bf16<<<(M_ * DM + 255) / 256, 256, 0, stream>>>(tmpf, tbf, M_ * DM);
  gemm16(tbf, wb_ow2, out_b2, preds_out, M_, ODIM, DM, 0);
  gemm16(hbf, wb_owr, out_br, preds_out, M_, ODIM, DM, 1);
}
